// TemporalDownScaleAttention_44710609552109
// MI455X (gfx1250) — matrix operand order repaired and checked by reading, not compiled, not run on hardware
//
#include <hip/hip_runtime.h>

typedef __bf16 bf16;
typedef bf16  v16bf __attribute__((ext_vector_type(16)));
typedef bf16  v8bf  __attribute__((ext_vector_type(8)));
typedef float v8f   __attribute__((ext_vector_type(8)));

#define DIM     512
#define TF      4
#define GT      16            // groups per workgroup
#define ROWS    64            // GT*TF token rows per workgroup
#define S       520           // padded LDS row stride (bf16 elems), keeps 16B align
#define KSTEPS  16            // 512 / 32
#define NTPW    4             // N-tiles per wave (32 tiles / 8 waves)

// ---- WMMA fragment loaders -------------------------------------------------
// A-matrix 16x32 bf16 (ISA 7.12.2): lane<16 -> row=lane, elems[0..7]=K kb..kb+7,
// elems[8..15]=K kb+16..kb+23 ; lane>=16 -> row=lane-16, K offsets +8.
__device__ __forceinline__ v16bf load_a_frag(const bf16* base, int lane, int kb, int stride) {
    int row = lane & 15;
    int off = (lane >> 4) * 8;
    const bf16* p = base + row * stride + kb + off;
    v16bf a;
    ((v8bf*)&a)[0] = *(const v8bf*)(p);
    ((v8bf*)&a)[1] = *(const v8bf*)(p + 16);
    return a;
}

// B-matrix 32x16 bf16 (KxN), B(k,n)=W[n,k] for e @ W^T. Per ISA B layout:
// lanes 0-15 hold K=kb..kb+7,kb+16..kb+23 for col n=lane; lanes 16-31 K offsets +8.
__device__ __forceinline__ v16bf load_b_frag(const bf16* __restrict__ w, int lane, int nbase, int kb) {
    int n    = lane & 15;
    int koff = kb + ((lane >> 4) * 8);
    const bf16* p = w + (nbase + n) * DIM + koff;
    v16bf b;
    ((v8bf*)&b)[0] = *(const v8bf*)(p);
    ((v8bf*)&b)[1] = *(const v8bf*)(p + 16);
    return b;
}

__device__ __forceinline__ v8f wmma_bf16(v16bf a, v16bf b, v8f c) {
    return __builtin_amdgcn_wmma_f32_16x16x32_bf16(false, a, false, b, (short)0, c, false, false);
}

// ---- prep: convert fp32 weights to bf16 in scratch -------------------------
// ws layout (bf16 elems): [Wq 512x512 | Wk | Wv | Wout]  (in_proj_w is Wq;Wk;Wv stacked)
__global__ __launch_bounds__(256) void prep_weights(const float* __restrict__ inw,
                                                    const float* __restrict__ outw,
                                                    bf16* __restrict__ ws) {
    int i = blockIdx.x * 256 + threadIdx.x;       // 1,048,576 total
    if (i < 786432)       ws[i] = (bf16)inw[i];
    else                  ws[i] = (bf16)outw[i - 786432];
}

// ---- fused kernel ----------------------------------------------------------
__global__ __launch_bounds__(256) void fused_group_attn(
    const float* __restrict__ x,        // [16384*4, 512]
    const float* __restrict__ pos,      // [4, 512]
    const float* __restrict__ bin,      // [1536] = bq|bk|bv
    const float* __restrict__ bo,       // [512]
    const bf16*  __restrict__ wq,
    const bf16*  __restrict__ wk,
    const bf16*  __restrict__ wv,
    const bf16*  __restrict__ wo,
    float* __restrict__ out)            // [16384, 512]
{
    __shared__ bf16  sE[ROWS * S];          // e = x + pos (bf16)
    __shared__ bf16  sK[ROWS * S];          // kh (bf16)
    __shared__ bf16  sQ[GT * S];            // q = mean(e)
    __shared__ float sQH[GT * DIM];         // qh (f32)
    __shared__ float sATT[GT * 8 * TF];     // softmax weights
    __shared__ bf16  sCTX[GT * S];          // ctx (bf16)

    const int  tid  = threadIdx.x;
    const int  wave = tid >> 5;
    const int  lane = tid & 31;
    const long rowBase = (long)blockIdx.x * ROWS;   // token row base
    const long gBase   = (long)blockIdx.x * GT;     // group base

    // ---------- Phase 1: e = x + pos_emb -> LDS bf16 ----------
    {
        const int c  = (tid & 127) * 4;   // 128 threads cover one row of 512
        const int r0 = tid >> 7;          // 0..1
        #pragma unroll 4
        for (int i = 0; i < 32; ++i) {
            int r = r0 + i * 2;
            float4 xv = *(const float4*)(x + (rowBase + r) * DIM + c);
            float4 pv = *(const float4*)(pos + (r & 3) * DIM + c);
            bf16* dst = &sE[r * S + c];
            dst[0] = (bf16)(xv.x + pv.x);
            dst[1] = (bf16)(xv.y + pv.y);
            dst[2] = (bf16)(xv.z + pv.z);
            dst[3] = (bf16)(xv.w + pv.w);
        }
    }
    __syncthreads();

    // ---------- Phase 1b: q = mean over 4 tokens ----------
    for (int i = tid; i < GT * DIM; i += 256) {
        int g = i >> 9, c = i & 511;
        float s = (float)sE[(g * 4 + 0) * S + c] + (float)sE[(g * 4 + 1) * S + c]
                + (float)sE[(g * 4 + 2) * S + c] + (float)sE[(g * 4 + 3) * S + c];
        sQ[g * S + c] = (bf16)(s * 0.25f);
    }
    __syncthreads();

    // ---------- Phase 2: kh = e@Wk^T + bk, qh = q@Wq^T + bq ----------
    #pragma unroll 1
    for (int i = 0; i < NTPW; ++i) {
        const int nt    = wave * NTPW + i;
        const int nbase = nt * 16;
        const int ncol  = nbase + (lane & 15);
        const float bkv = bin[512 + ncol];
        const float bqv = bin[ncol];
        v8f accK[4] = {};
        v8f accQ    = {};
        #pragma unroll 1
        for (int k = 0; k < KSTEPS; ++k) {
            const int kb = k * 32;
            v16bf bK = load_b_frag(wk, lane, nbase, kb);
            v16bf bQ = load_b_frag(wq, lane, nbase, kb);
            v16bf aq = load_a_frag(sQ, lane, kb, S);
            accQ = wmma_bf16(aq, bQ, accQ);
            #pragma unroll
            for (int m = 0; m < 4; ++m) {
                v16bf a = load_a_frag(sE + m * 16 * S, lane, kb, S);
                accK[m] = wmma_bf16(a, bK, accK[m]);
            }
        }
        const int rh = (lane >> 4) * 8;   // row-half offset from C/D layout
        #pragma unroll
        for (int m = 0; m < 4; ++m)
            #pragma unroll
            for (int j = 0; j < 8; ++j)
                sK[(m * 16 + rh + j) * S + ncol] = (bf16)(accK[m][j] + bkv);
        #pragma unroll
        for (int j = 0; j < 8; ++j)
            sQH[(rh + j) * DIM + ncol] = accQ[j] + bqv;
    }
    __syncthreads();

    // ---------- Phase 3: per-(group,head) softmax over 4 tokens ----------
    if (tid < GT * 8) {
        const int g = tid >> 3, h = tid & 7;
        const float* qp = &sQH[g * DIM + h * 64];
        float sc[TF];
        #pragma unroll
        for (int t = 0; t < TF; ++t) {
            const bf16* kp = &sK[(g * 4 + t) * S + h * 64];
            float s = 0.f;
            for (int c = 0; c < 64; ++c) s += qp[c] * (float)kp[c];
            sc[t] = s * 0.125f;           // 1/sqrt(64)
        }
        float mx = fmaxf(fmaxf(sc[0], sc[1]), fmaxf(sc[2], sc[3]));
        float sum = 0.f;
        #pragma unroll
        for (int t = 0; t < TF; ++t) { sc[t] = __expf(sc[t] - mx); sum += sc[t]; }
        const float inv = 1.f / sum;
        #pragma unroll
        for (int t = 0; t < TF; ++t) sATT[(g * 8 + h) * TF + t] = sc[t] * inv;
    }
    __syncthreads();

    // ---------- Phase 4: vh = e@Wv^T + bv with fused ctx = sum_t attn*vh ----------
    // In the D layout each lane's 8 regs = 2 complete groups (4 tokens each) of one column.
    #pragma unroll 1
    for (int i = 0; i < NTPW; ++i) {
        const int nt    = wave * NTPW + i;
        const int nbase = nt * 16;
        const int ncol  = nbase + (lane & 15);
        const int h     = nt >> 2;        // 16 cols of a tile stay inside one head (HD=64)
        const float bvv = bin[1024 + ncol];
        v8f acc[4] = {};
        #pragma unroll 1
        for (int k = 0; k < KSTEPS; ++k) {
            const int kb = k * 32;
            v16bf bV = load_b_frag(wv, lane, nbase, kb);
            #pragma unroll
            for (int m = 0; m < 4; ++m) {
                v16bf a = load_a_frag(sE + m * 16 * S, lane, kb, S);
                acc[m] = wmma_bf16(a, bV, acc[m]);
            }
        }
        #pragma unroll
        for (int m = 0; m < 4; ++m) {
            const int g0 = m * 4 + (lane >> 4) * 2;   // first group in this lane-half
            const float* a0 = &sATT[(g0 * 8 + h) * TF];
            const float* a1 = &sATT[((g0 + 1) * 8 + h) * TF];
            float c0 = bvv, c1 = bvv;
            #pragma unroll
            for (int t = 0; t < TF; ++t) {
                c0 += a0[t] * acc[m][t];
                c1 += a1[t] * acc[m][4 + t];
            }
            sCTX[g0 * S + ncol]       = (bf16)c0;
            sCTX[(g0 + 1) * S + ncol] = (bf16)c1;
        }
    }
    __syncthreads();

    // ---------- Phase 5: out = ctx@Wout^T + bout ----------
    #pragma unroll 1
    for (int i = 0; i < NTPW; ++i) {
        const int nt    = wave * NTPW + i;
        const int nbase = nt * 16;
        const int ncol  = nbase + (lane & 15);
        const float bov = bo[ncol];
        v8f acc = {};
        #pragma unroll 1
        for (int k = 0; k < KSTEPS; ++k) {
            const int kb = k * 32;
            v16bf b = load_b_frag(wo, lane, nbase, kb);
            v16bf a = load_a_frag(sCTX, lane, kb, S);
            acc = wmma_bf16(a, b, acc);
        }
        const int rh = (lane >> 4) * 8;
        #pragma unroll
        for (int j = 0; j < 8; ++j)
            out[(gBase + rh + j) * DIM + ncol] = acc[j] + bov;
    }
}

extern "C" void kernel_launch(void* const* d_in, const int* in_sizes, int n_in,
                              void* d_out, int out_size, void* d_ws, size_t ws_size,
                              hipStream_t stream) {
    const float* x    = (const float*)d_in[0];   // [8, 8192, 512]
    const float* pos  = (const float*)d_in[1];   // [4, 512]
    const float* inw  = (const float*)d_in[2];   // [1536, 512]
    const float* inb  = (const float*)d_in[3];   // [1536]
    const float* outw = (const float*)d_in[4];   // [512, 512]
    const float* outb = (const float*)d_in[5];   // [512]
    float* out = (float*)d_out;                  // [16384, 512]

    bf16* ws = (bf16*)d_ws;                      // needs 2 MB
    const bf16* wq = ws;
    const bf16* wk = ws + 512 * 512;
    const bf16* wv = ws + 2 * 512 * 512;
    const bf16* wo = ws + 3 * 512 * 512;

    prep_weights<<<4096, 256, 0, stream>>>(inw, outw, ws);

    const int numGroups = 16384;
    fused_group_attn<<<numGroups / GT, 256, 0, stream>>>(
        x, pos, inb, outb, wq, wk, wv, wo, out);
}
